// SwarmVAE_70497593196925
// MI455X (gfx1250) — compile-verified, matrix operand order checked by reading, not run on hardware
//
#include <hip/hip_runtime.h>

// ---------------- constants ----------------
#define BATCH 32768

typedef __attribute__((ext_vector_type(16))) _Float16 v16h;
typedef __attribute__((ext_vector_type(8)))  _Float16 v8h;
typedef __attribute__((ext_vector_type(8)))  float    v8f;
typedef __attribute__((ext_vector_type(4)))  int      v4i;

#if __has_builtin(__builtin_amdgcn_global_load_async_to_lds_b128) && \
    __has_builtin(__builtin_amdgcn_s_wait_asynccnt)
#define USE_ASYNC_LDS 1
#else
#define USE_ASYNC_LDS 0
#endif

// ---------------- WMMA helpers ----------------
// A-fragment (16x32 f16, MxK): lane l holds row m=l&15; K chunks [kb..kb+7] and
// [kb+16..kb+23] with kb = (l>=16)*8.  B-fragment mirrors it (n in place of m),
// so feeding the same loader with a row-major (N x K) weight matrix yields
// D = A * W^T (the reference's `x @ W.T` convention).
__device__ __forceinline__ v16h load_frag(const _Float16* base, int row0, int ld, int k0) {
    const int lane = threadIdx.x & 31;
    const int r    = lane & 15;
    const int kb   = (lane >> 4) << 3;
    const _Float16* p = base + (size_t)(row0 + r) * ld + (k0 + kb);
    v8h a = *(const v8h*)p;
    v8h b = *(const v8h*)(p + 16);
    return __builtin_shufflevector(a, b, 0,1,2,3,4,5,6,7,8,9,10,11,12,13,14,15);
}

__device__ __forceinline__ v8f wmma_f16(v16h a, v16h b, v8f c) {
    return __builtin_amdgcn_wmma_f32_16x16x32_f16(false, a, false, b, (short)0, c, false, false);
}

// Single M-row tile x NT col tiles.
template<int NT>
__device__ __forceinline__ void gemm_rb(const _Float16* A, int am0, int lda,
                                        const _Float16* W, int wn0, int ldw,
                                        int K, v8f* acc) {
    for (int k = 0; k < K; k += 32) {
        v16h af = load_frag(A, am0, lda, k);
#pragma unroll
        for (int i = 0; i < NT; ++i) {
            v16h bf = load_frag(W, wn0 + i * 16, ldw, k);
            acc[i] = wmma_f16(af, bf, acc[i]);
        }
    }
}

// Dual M-row tiles sharing each B-fragment: halves weight (L2) traffic per WMMA.
template<int NT>
__device__ __forceinline__ void gemm_rb2(const _Float16* A, int am0, int am1, int lda,
                                         const _Float16* W, int wn0, int ldw,
                                         int K, v8f* acc0, v8f* acc1) {
    for (int k = 0; k < K; k += 32) {
        v16h a0 = load_frag(A, am0, lda, k);
        v16h a1 = load_frag(A, am1, lda, k);
#pragma unroll
        for (int i = 0; i < NT; ++i) {
            v16h bf = load_frag(W, wn0 + i * 16, ldw, k);
            acc0[i] = wmma_f16(a0, bf, acc0[i]);
            acc1[i] = wmma_f16(a1, bf, acc1[i]);
        }
    }
}

// C/D layout: VGPR r -> M = r + 8*(lane>=16), N = lane&15
__device__ __forceinline__ void store_tile_f16(_Float16* dst, int row0, int ld, int col0,
                                               v8f acc, const float* bias, int bcol, bool relu) {
    const int lane = threadIdx.x & 31;
    const int n    = lane & 15;
    const int mo   = (lane >> 4) << 3;
    const float bv = bias[bcol + n];
#pragma unroll
    for (int r = 0; r < 8; ++r) {
        float v = acc[r] + bv;
        if (relu) v = v > 0.f ? v : 0.f;
        dst[(size_t)(row0 + mo + r) * ld + col0 + n] = (_Float16)v;
    }
}

// ---------------- weight fp32 -> f16 ----------------
__global__ void cvt_kernel(const float* __restrict__ s, _Float16* __restrict__ d, int n) {
    int i = blockIdx.x * 256 + threadIdx.x;
    if (i < n) d[i] = (_Float16)s[i];
}

// ---------------- fused front: encoder + heads + sample + attention + consensus ----
struct FrontArgs {
    const float* x; const float* eps;
    const _Float16 *W1, *W2, *muW, *lvW, *Wqkv, *Wo, *Wc;
    const float *b1, *b2, *mub, *lvb, *bqkv, *bo, *bc;
    _Float16* Zc;   // (B, 5, 64) f16: slot0 = consensus, slots 1..4 = agents
};

__launch_bounds__(256)
__global__ void front_kernel(FrontArgs A) {
    __shared__ __align__(16) _Float16 xt[32 * 512];     // x tile, later h2 tile
    __shared__ __align__(16) _Float16 h1[32 * 512];
    __shared__ __align__(16) _Float16 zt[128 * 64];     // z (agents), later ctx
    __shared__ __align__(16) _Float16 qkvt[128 * 192];
    __shared__ __align__(16) _Float16 za[128 * 64];     // z_att, viewed as (32,256)

    const int tid  = threadIdx.x;
    const int w    = tid >> 5;
    const int row0 = blockIdx.x * 32;

    // stage x tile (32x512) fp32 -> f16
    for (int i = tid; i < 32 * 512; i += 256)
        xt[i] = (_Float16)A.x[(size_t)row0 * 512 + i];
    __syncthreads();

    // ---- encoder layer 1: h1 = relu(x @ W1^T + b1); wave w -> cols [w*64, w*64+64) ----
    {
        const int c0 = w * 64;
        v8f a0[4] = {}, a1[4] = {};
        gemm_rb2<4>(xt, 0, 16, 512, A.W1, c0, 512, 512, a0, a1);
#pragma unroll
        for (int i = 0; i < 4; ++i) {
            store_tile_f16(h1, 0,  512, c0 + i * 16, a0[i], A.b1, c0 + i * 16, true);
            store_tile_f16(h1, 16, 512, c0 + i * 16, a1[i], A.b1, c0 + i * 16, true);
        }
    }
    __syncthreads();

    // ---- encoder layer 2: h2 = relu(h1 @ W2^T + b2) -> overwrite xt ----
    {
        const int c0 = w * 64;
        v8f a0[4] = {}, a1[4] = {};
        gemm_rb2<4>(h1, 0, 16, 512, A.W2, c0, 512, 512, a0, a1);
#pragma unroll
        for (int i = 0; i < 4; ++i) {
            store_tile_f16(xt, 0,  512, c0 + i * 16, a0[i], A.b2, c0 + i * 16, true);
            store_tile_f16(xt, 16, 512, c0 + i * 16, a1[i], A.b2, c0 + i * 16, true);
        }
    }
    __syncthreads();

    // ---- heads + reparameterization: z = mu + eps * exp(0.5*lv) * T ----
    {
        const float T[4]  = {1.5f, 0.5f, 1.0f, 0.8f};
        const float LT[4] = {0.4054651f, -0.6931472f, 0.0f, -0.2231436f};
        const int c0 = w * 32;          // two col tiles per wave over N=256
        v8f mu0[2] = {}, mu1[2] = {}, lv0[2] = {}, lv1[2] = {};
        gemm_rb2<2>(xt, 0, 16, 512, A.muW, c0, 512, 512, mu0, mu1);
        gemm_rb2<2>(xt, 0, 16, 512, A.lvW, c0, 512, 512, lv0, lv1);
        const int lane = tid & 31, n = lane & 15, mo = (lane >> 4) << 3;
#pragma unroll
        for (int i = 0; i < 2; ++i) {
            const int col = c0 + i * 16 + n;
            const int a = col >> 6, l = col & 63;
            const float mb = A.mub[col];
            const float lb = A.lvb[col] + LT[a];
#pragma unroll
            for (int rt = 0; rt < 2; ++rt) {
                const v8f* mm = rt ? mu1 : mu0;
                const v8f* ll = rt ? lv1 : lv0;
#pragma unroll
                for (int r = 0; r < 8; ++r) {
                    const int m = rt * 16 + mo + r;
                    float mu = mm[i][r] + mb;
                    float lv = ll[i][r] + lb;
                    float z  = mu + A.eps[(size_t)(row0 + m) * 256 + col] *
                                    __expf(0.5f * lv) * T[a];
                    zt[(m * 4 + a) * 64 + l] = (_Float16)z;
                    A.Zc[(size_t)(row0 + m) * 320 + (a + 1) * 64 + l] = (_Float16)z;
                }
            }
        }
    }
    __syncthreads();

    // ---- qkv = z @ Wqkv^T + bqkv : M=128, N=192, K=64 ----
    {
        const int am0 = (w & 3) * 32, am1 = am0 + 16;
        const int c0  = (w >> 2) * 96;    // 6 col tiles each half
        v8f a0[6] = {}, a1[6] = {};
        gemm_rb2<6>(zt, am0, am1, 64, A.Wqkv, c0, 64, 64, a0, a1);
#pragma unroll
        for (int i = 0; i < 6; ++i) {
            store_tile_f16(qkvt, am0, 192, c0 + i * 16, a0[i], A.bqkv, c0 + i * 16, false);
            store_tile_f16(qkvt, am1, 192, c0 + i * 16, a1[i], A.bqkv, c0 + i * 16, false);
        }
    }
    __syncthreads();

    // ---- tiny attention over A=4 agents, NH=4 heads, HD=16; ctx -> zt ----
    for (int it = tid; it < 512; it += 256) {
        const int bi = it >> 4, hi = (it >> 2) & 3, qi = it & 3;
        const _Float16* qp = &qkvt[(bi * 4 + qi) * 192 + hi * 16];
        float s[4], mx = -1e30f;
#pragma unroll
        for (int j = 0; j < 4; ++j) {
            const _Float16* kp = &qkvt[(bi * 4 + j) * 192 + 64 + hi * 16];
            float d = 0.f;
#pragma unroll
            for (int d0 = 0; d0 < 16; ++d0) d += (float)qp[d0] * (float)kp[d0];
            s[j] = d * 0.25f;                 // 1/sqrt(16)
            mx = s[j] > mx ? s[j] : mx;
        }
        float ssum = 0.f;
#pragma unroll
        for (int j = 0; j < 4; ++j) { s[j] = __expf(s[j] - mx); ssum += s[j]; }
        const float inv = 1.f / ssum;
        float ctx[16];
#pragma unroll
        for (int d0 = 0; d0 < 16; ++d0) ctx[d0] = 0.f;
#pragma unroll
        for (int j = 0; j < 4; ++j) {
            const _Float16* vp = &qkvt[(bi * 4 + j) * 192 + 128 + hi * 16];
            const float p = s[j] * inv;
#pragma unroll
            for (int d0 = 0; d0 < 16; ++d0) ctx[d0] += p * (float)vp[d0];
        }
#pragma unroll
        for (int d0 = 0; d0 < 16; ++d0)
            zt[(bi * 4 + qi) * 64 + hi * 16 + d0] = (_Float16)ctx[d0];
    }
    __syncthreads();

    // ---- z_att = ctx @ Wo^T + bo : M=128, N=64, K=64 ----
    {
        const int am0 = (w & 3) * 32, am1 = am0 + 16;
        const int c0  = (w >> 2) * 32;    // 2 col tiles each half
        v8f a0[2] = {}, a1[2] = {};
        gemm_rb2<2>(zt, am0, am1, 64, A.Wo, c0, 64, 64, a0, a1);
#pragma unroll
        for (int i = 0; i < 2; ++i) {
            store_tile_f16(za, am0, 64, c0 + i * 16, a0[i], A.bo, c0 + i * 16, false);
            store_tile_f16(za, am1, 64, c0 + i * 16, a1[i], A.bo, c0 + i * 16, false);
        }
    }
    __syncthreads();

    // ---- consensus: (32,256) @ Wc^T + bc -> Zc slot 0 ----
    {
        const int tm = (w & 1) * 16;
        const int tc = (w >> 1) * 16;
        v8f acc[1] = {};
        gemm_rb<1>(za, tm, 256, A.Wc, tc, 256, 256, acc);
        const int lane = tid & 31, n = lane & 15, mo = (lane >> 4) << 3;
        const float bv = A.bc[tc + n];
#pragma unroll
        for (int r = 0; r < 8; ++r) {
            const int m = tm + mo + r;
            A.Zc[(size_t)(row0 + m) * 320 + tc + n] = (_Float16)(acc[0][r] + bv);
        }
    }
}

// ---------------- fused 3-layer decoder over (5B, 64) latents ----------------
struct DecArgs {
    const _Float16* Zc;
    const _Float16 *Wd1, *Wd2, *Wd3;
    const float *bd1, *bd2, *bd3;
    float* out;             // (5B, 512) == (B, A+1, D)
};

__launch_bounds__(256)
__global__ void decoder_kernel(DecArgs A) {
    __shared__ __align__(16) _Float16 zt[64 * 64];
    __shared__ __align__(16) _Float16 d1[64 * 512];
    __shared__ __align__(16) _Float16 d2[64 * 512];

    const int tid  = threadIdx.x;
    const int w    = tid >> 5;
    const size_t row0 = (size_t)blockIdx.x * 64;

    // stage 64x64 latent tile (8 KB, contiguous f16) — async-to-LDS on gfx1250
#if USE_ASYNC_LDS
    for (int c = tid; c < 512; c += 256)
        __builtin_amdgcn_global_load_async_to_lds_b128(
            (__attribute__((address_space(1))) v4i*)(A.Zc + row0 * 64 + (size_t)c * 8),
            (__attribute__((address_space(3))) v4i*)(zt + c * 8), 0, 0);
    __builtin_amdgcn_s_wait_asynccnt(0);
#else
    for (int c = tid; c < 512; c += 256)
        ((v8h*)zt)[c] = ((const v8h*)(A.Zc + row0 * 64))[c];
#endif
    __syncthreads();

    const int tm = (w & 1) * 16;        // row tiles tm and tm+32
    const int cb = (w >> 1) * 128;      // 8 col tiles per wave

    {   // layer 1: relu(z @ Wd1^T + bd1), K=64
        v8f a0[8] = {}, a1[8] = {};
        gemm_rb2<8>(zt, tm, tm + 32, 64, A.Wd1, cb, 64, 64, a0, a1);
#pragma unroll
        for (int i = 0; i < 8; ++i) {
            store_tile_f16(d1, tm,      512, cb + i * 16, a0[i], A.bd1, cb + i * 16, true);
            store_tile_f16(d1, tm + 32, 512, cb + i * 16, a1[i], A.bd1, cb + i * 16, true);
        }
    }
    __syncthreads();
    {   // layer 2: relu(d1 @ Wd2^T + bd2), K=512
        v8f a0[8] = {}, a1[8] = {};
        gemm_rb2<8>(d1, tm, tm + 32, 512, A.Wd2, cb, 512, 512, a0, a1);
#pragma unroll
        for (int i = 0; i < 8; ++i) {
            store_tile_f16(d2, tm,      512, cb + i * 16, a0[i], A.bd2, cb + i * 16, true);
            store_tile_f16(d2, tm + 32, 512, cb + i * 16, a1[i], A.bd2, cb + i * 16, true);
        }
    }
    __syncthreads();
    {   // layer 3: d2 @ Wd3^T + bd3 -> fp32 out
        v8f a0[8] = {}, a1[8] = {};
        gemm_rb2<8>(d2, tm, tm + 32, 512, A.Wd3, cb, 512, 512, a0, a1);
        const int lane = tid & 31, n = lane & 15, mo = (lane >> 4) << 3;
#pragma unroll
        for (int i = 0; i < 8; ++i) {
            const int col = cb + i * 16 + n;
            const float bv = A.bd3[col];
#pragma unroll
            for (int r = 0; r < 8; ++r) {
                A.out[(row0 + tm + mo + r) * 512 + col]      = a0[i][r] + bv;
                A.out[(row0 + tm + 32 + mo + r) * 512 + col] = a1[i][r] + bv;
            }
        }
    }
}

// ---------------- host-side launcher ----------------
extern "C" void kernel_launch(void* const* d_in, const int* in_sizes, int n_in,
                              void* d_out, int out_size, void* d_ws, size_t ws_size,
                              hipStream_t stream) {
    const float* x    = (const float*)d_in[0];
    const float* eps  = (const float*)d_in[1];
    const float* W1   = (const float*)d_in[2];
    const float* b1   = (const float*)d_in[3];
    const float* W2   = (const float*)d_in[4];
    const float* b2   = (const float*)d_in[5];
    const float* muW  = (const float*)d_in[6];
    const float* mub  = (const float*)d_in[7];
    const float* lvW  = (const float*)d_in[8];
    const float* lvb  = (const float*)d_in[9];
    const float* Wqkv = (const float*)d_in[10];
    const float* bqkv = (const float*)d_in[11];
    const float* Wo   = (const float*)d_in[12];
    const float* bo   = (const float*)d_in[13];
    const float* Wc   = (const float*)d_in[14];
    const float* bc   = (const float*)d_in[15];
    const float* Wd1  = (const float*)d_in[16];
    const float* bd1  = (const float*)d_in[17];
    const float* Wd2  = (const float*)d_in[18];
    const float* bd2  = (const float*)d_in[19];
    const float* Wd3  = (const float*)d_in[20];
    const float* bd3  = (const float*)d_in[21];
    float* out = (float*)d_out;

    // carve workspace (256B aligned)
    char* p = (char*)d_ws;
    auto carve = [&](size_t nel) {
        _Float16* r = (_Float16*)p;
        p += ((nel * 2 + 255) & ~(size_t)255);
        return r;
    };
    _Float16* W1h   = carve(512 * 512);
    _Float16* W2h   = carve(512 * 512);
    _Float16* muWh  = carve(256 * 512);
    _Float16* lvWh  = carve(256 * 512);
    _Float16* Wqkvh = carve(192 * 64);
    _Float16* Woh   = carve(64 * 64);
    _Float16* Wch   = carve(64 * 256);
    _Float16* Wd1h  = carve(512 * 64);
    _Float16* Wd2h  = carve(512 * 512);
    _Float16* Wd3h  = carve(512 * 512);
    _Float16* Zc    = carve((size_t)BATCH * 320);

    struct { const float* s; _Float16* d; int n; } cv[10] = {
        {W1, W1h, 512 * 512}, {W2, W2h, 512 * 512},
        {muW, muWh, 256 * 512}, {lvW, lvWh, 256 * 512},
        {Wqkv, Wqkvh, 192 * 64}, {Wo, Woh, 64 * 64},
        {Wc, Wch, 64 * 256}, {Wd1, Wd1h, 512 * 64},
        {Wd2, Wd2h, 512 * 512}, {Wd3, Wd3h, 512 * 512},
    };
    for (int i = 0; i < 10; ++i)
        cvt_kernel<<<(cv[i].n + 255) / 256, 256, 0, stream>>>(cv[i].s, cv[i].d, cv[i].n);

    FrontArgs fa;
    fa.x = x; fa.eps = eps;
    fa.W1 = W1h; fa.W2 = W2h; fa.muW = muWh; fa.lvW = lvWh;
    fa.Wqkv = Wqkvh; fa.Wo = Woh; fa.Wc = Wch;
    fa.b1 = b1; fa.b2 = b2; fa.mub = mub; fa.lvb = lvb;
    fa.bqkv = bqkv; fa.bo = bo; fa.bc = bc;
    fa.Zc = Zc;
    front_kernel<<<BATCH / 32, 256, 0, stream>>>(fa);

    DecArgs da;
    da.Zc = Zc; da.Wd1 = Wd1h; da.Wd2 = Wd2h; da.Wd3 = Wd3h;
    da.bd1 = bd1; da.bd2 = bd2; da.bd3 = bd3;
    da.out = out;
    decoder_kernel<<<(BATCH * 5) / 64, 256, 0, stream>>>(da);
}
